// RotarySelfAttention_18150531793641
// MI455X (gfx1250) — matrix operand order repaired and checked by reading, not compiled, not run on hardware
//
#include <hip/hip_runtime.h>

// ---------------------------------------------------------------------------
// RotarySelfAttention for MI455X (gfx1250, wave32, WMMA bf16 16x16x32)
//   B=2, S=2048, DIM=1536, H=16, HD=96, NUM_PAIRS=48
// Pipeline:
//   1) convert x, qkv_w, proj_w  f32 -> bf16
//   2) QKV GEMM (WMMA bf16, f32 acc, 32x64 wave tile, double-buffered A)
//      -> Q,K as [B,H,S,HD], V as [B,H,HD,S] (pre-transposed)
//   3) RoPE on q,k  (folds the effective 1/HD logit scale into q)
//   4) Flash attention (WMMA QK^T and PV, online softmax, f32 stats)
//      256 threads / 8 waves per block; K/V tiles shared via LDS
//   5) Proj GEMM (WMMA, 32x64 wave tile) + bias -> f32 output [B,S,DIM]
// ---------------------------------------------------------------------------

#define B_ 2
#define S_ 2048
#define DIM_ 1536
#define H_ 16
#define HD_ 96
#define NPAIR_ 48
#define QKVN_ (3 * DIM_)   // 4608
#define M_ (B_ * S_)       // 4096

typedef __bf16 bf16_t;
typedef unsigned short ushort_t;
typedef bf16_t v16bf __attribute__((ext_vector_type(16)));
typedef bf16_t v8bf  __attribute__((ext_vector_type(8)));
typedef float  v8f   __attribute__((ext_vector_type(8)));

// ---------------- bf16 helpers (storage = ushort, RNE rounding) ------------
static __device__ __forceinline__ ushort_t f32_to_bf16(float f) {
    union { float f; unsigned int u; } c; c.f = f;
    unsigned int u = c.u + 0x7FFFu + ((c.u >> 16) & 1u);
    return (ushort_t)(u >> 16);
}
static __device__ __forceinline__ float bf16_to_f32(ushort_t h) {
    union { unsigned int u; float f; } c; c.u = ((unsigned int)h) << 16;
    return c.f;
}
static __device__ __forceinline__ v8bf load8(const ushort_t* p) {
    return *reinterpret_cast<const v8bf*>(p);
}
static __device__ __forceinline__ v8f zero8() {
    v8f z;
#pragma unroll
    for (int i = 0; i < 8; ++i) z[i] = 0.0f;
    return z;
}

// ---------------- WMMA fragment loaders (ISA 7.12.2 layouts) ---------------
// A-matrix 16x32 bf16: lane<16 -> M=lane, K={k0..k0+7, k0+16..k0+23}
//                      lane>=16 -> M=lane-16, K={k0+8..k0+15, k0+24..k0+31}
static __device__ __forceinline__ v16bf frag_a(const ushort_t* base, int ld,
                                               int row0, int k0, int lane) {
    int m = lane & 15, hi = lane >> 4;
    const ushort_t* r = base + (size_t)(row0 + m) * ld + k0 + hi * 8;
    v8bf c0 = load8(r);
    v8bf c1 = load8(r + 16);
    v16bf a;
#pragma unroll
    for (int i = 0; i < 8; ++i) { a[i] = c0[i]; a[i + 8] = c1[i]; }
    return a;
}
// B-matrix 32x16 bf16 with Bmat[k][n] = W[n][k] (W row-major [N][K]):
// lane<16 -> N=lane, K={k0..k0+7, k0+16..k0+23} ; lane>=16 -> N=lane-16, K={k0+8..k0+15, k0+24..k0+31}
static __device__ __forceinline__ v16bf frag_b(const ushort_t* base, int ld,
                                               int n0, int k0, int lane) {
    int n = lane & 15, hi = lane >> 4;
    const ushort_t* r = base + (size_t)(n0 + n) * ld + k0 + hi * 8;
    v8bf c0 = load8(r);
    v8bf c1 = load8(r + 16);
    v16bf b;
#pragma unroll
    for (int i = 0; i < 8; ++i) { b[i] = c0[i]; b[i + 8] = c1[i]; }
    return b;
}
static __device__ __forceinline__ v8f wmma_bf16(v16bf a, v16bf b, v8f c) {
    return __builtin_amdgcn_wmma_f32_16x16x32_bf16(
        /*neg_a=*/false, a, /*neg_b=*/false, b,
        /*c_mod=*/(short)0, c, /*reuse_a=*/false, /*reuse_b=*/false);
}

// ---------------- Kernel 1: f32 -> bf16 conversion -------------------------
__global__ void convert_f32_bf16(const float* __restrict__ s,
                                 ushort_t* __restrict__ d, int n) {
    int i = blockIdx.x * blockDim.x + threadIdx.x;
    if (i < n) d[i] = f32_to_bf16(s[i]);
}

// ---------------- Kernel 2: QKV GEMM ---------------------------------------
// C[M, 3*DIM] = xb[M,DIM] @ qkv_wb[3*DIM,DIM]^T + qkv_b
// Wave tile 32x64 (2 M-subtiles x 4 N-subtiles), A double-buffered so B-frag
// loads + next-A loads are all in flight before the 8-WMMA burst.
// Scatter: Q,K parts -> [part][B][H][S][HD] ; V part -> [2][B][H][HD][S]
__global__ void __launch_bounds__(128) qkv_gemm(const ushort_t* __restrict__ xb,
                                                const ushort_t* __restrict__ wb,
                                                const float* __restrict__ bias,
                                                ushort_t* __restrict__ qkv) {
    int mt = blockIdx.x * 32;
    int wave = threadIdx.x >> 5, lane = threadIdx.x & 31;
    int nbase = blockIdx.y * 256 + wave * 64;

    v8f acc0[4], acc1[4];
#pragma unroll
    for (int t = 0; t < 4; ++t) { acc0[t] = zero8(); acc1[t] = zero8(); }

    v16bf a0 = frag_a(xb, DIM_, mt, 0, lane);
    v16bf a1 = frag_a(xb, DIM_, mt + 16, 0, lane);
    for (int k0 = 0; k0 < DIM_; k0 += 32) {
        // all four B-frags in distinct registers + next A-frags: independent
        // loads in flight before the WMMA burst
        v16bf b0 = frag_b(wb, DIM_, nbase,      k0, lane);
        v16bf b1 = frag_b(wb, DIM_, nbase + 16, k0, lane);
        v16bf b2 = frag_b(wb, DIM_, nbase + 32, k0, lane);
        v16bf b3 = frag_b(wb, DIM_, nbase + 48, k0, lane);
        int kn = (k0 + 32 < DIM_) ? (k0 + 32) : 0;   // last-iter reload is harmless
        v16bf a0n = frag_a(xb, DIM_, mt,      kn, lane);
        v16bf a1n = frag_a(xb, DIM_, mt + 16, kn, lane);
        acc0[0] = wmma_bf16(a0, b0, acc0[0]);
        acc1[0] = wmma_bf16(a1, b0, acc1[0]);
        acc0[1] = wmma_bf16(a0, b1, acc0[1]);
        acc1[1] = wmma_bf16(a1, b1, acc1[1]);
        acc0[2] = wmma_bf16(a0, b2, acc0[2]);
        acc1[2] = wmma_bf16(a1, b2, acc1[2]);
        acc0[3] = wmma_bf16(a0, b3, acc0[3]);
        acc1[3] = wmma_bf16(a1, b3, acc1[3]);
        a0 = a0n; a1 = a1n;
    }
    // Epilogue: C-frag lane = N, vgpr v -> M = v + hi*8
    int nl = lane & 15, hi = lane >> 4;
#pragma unroll
    for (int ms = 0; ms < 2; ++ms) {
        v8f* acc = ms ? acc1 : acc0;
#pragma unroll
        for (int t = 0; t < 4; ++t) {
            int n = nbase + t * 16 + nl;
            float bs = bias[n];
            int part = n / DIM_, rem = n % DIM_;
            int head = rem / HD_, hd = rem % HD_;
#pragma unroll
            for (int v = 0; v < 8; ++v) {
                int m = mt + ms * 16 + v + hi * 8;
                int bb = m >> 11, tt = m & (S_ - 1);
                size_t idx;
                if (part == 2) {
                    // V transposed: [2*B*H*S*HD] + ((b*H+head)*HD+hd)*S + t
                    idx = (size_t)2 * B_ * H_ * S_ * HD_ +
                          (((size_t)bb * H_ + head) * HD_ + hd) * S_ + tt;
                } else {
                    idx = (((((size_t)part * B_ + bb) * H_ + head) * S_ + tt) * HD_) + hd;
                }
                qkv[idx] = f32_to_bf16(acc[t][v] + bs);
            }
        }
    }
}

// ---------------- Kernel 3: mixed 3D RoPE on q and k -----------------------
__global__ void rope_kernel(ushort_t* __restrict__ qkv,
                            const float* __restrict__ coords,
                            const float* __restrict__ mixf) {
    int idx = blockIdx.x * blockDim.x + threadIdx.x;
    const int total = B_ * H_ * S_ * NPAIR_;
    if (idx >= total) return;
    int p = idx % NPAIR_;
    int t = (idx / NPAIR_) % S_;
    int h = (idx / (NPAIR_ * S_)) % H_;
    int b = idx / (NPAIR_ * S_ * H_);

    const float* c  = coords + ((size_t)b * S_ + t) * 3;
    const float* mf = mixf + ((size_t)h * NPAIR_ + p) * 3;
    float ang = 6.283185307179586f * (c[0] * mf[0] + c[1] * mf[1] + c[2] * mf[2]);
    float sn, cs;
    __sincosf(ang, &sn, &cs);

    size_t base_q = ((((size_t)0 * B_ + b) * H_ + h) * S_ + t) * HD_;
    size_t base_k = base_q + (size_t)B_ * H_ * S_ * HD_;

    // q: fold effective logit scale (HD^-0.5 applied twice) = 1/HD into q
    const float qs = 1.0f / (float)HD_;
    float q1 = bf16_to_f32(qkv[base_q + p]);
    float q2 = bf16_to_f32(qkv[base_q + p + NPAIR_]);
    qkv[base_q + p]          = f32_to_bf16((q1 * cs - q2 * sn) * qs);
    qkv[base_q + p + NPAIR_] = f32_to_bf16((q2 * cs + q1 * sn) * qs);

    float k1 = bf16_to_f32(qkv[base_k + p]);
    float k2 = bf16_to_f32(qkv[base_k + p + NPAIR_]);
    qkv[base_k + p]          = f32_to_bf16(k1 * cs - k2 * sn);
    qkv[base_k + p + NPAIR_] = f32_to_bf16(k2 * cs + k1 * sn);
}

// ---------------- Kernel 4: flash attention --------------------------------
// Block = 256 threads (8 waves); each wave owns a 16-row q tile; the block
// shares one (b,h) and 32-key tiles staged in LDS.  V arrives pre-transposed
// ([B,H,HD,S]) so both K and V tiles stage with pure b128 vector loads.
__global__ void __launch_bounds__(256) flash_attn(const ushort_t* __restrict__ Q,
                                                  const ushort_t* __restrict__ K,
                                                  const ushort_t* __restrict__ V,
                                                  ushort_t* __restrict__ out) {
    __shared__ ushort_t Kt[32 * HD_];       // [key][hd]   6 KB
    __shared__ ushort_t Vt[HD_ * 32];       // [hd][key]   6 KB
    __shared__ ushort_t Pst[8][16 * 32];    // per-wave P staging  8 KB

    int bh   = blockIdx.x >> 4;             // S/128 = 16 q-blocks per (b,h)
    int qblk = blockIdx.x & 15;
    int wave = threadIdx.x >> 5, lane = threadIdx.x & 31;
    int q0 = qblk * 128 + wave * 16;

    const ushort_t* Qp = Q + (size_t)bh * S_ * HD_;   // [t][hd]
    const ushort_t* Kp = K + (size_t)bh * S_ * HD_;   // [t][hd]
    const ushort_t* Vg = V + (size_t)bh * HD_ * S_;   // [hd][t] (transposed)

    // Q fragments held for the whole loop (HD=96 -> 3 K-steps of 32)
    v16bf aQ[3];
#pragma unroll
    for (int j = 0; j < 3; ++j) aQ[j] = frag_a(Qp, HD_, q0, j * 32, lane);

    v8f o[6];
#pragma unroll
    for (int f = 0; f < 6; ++f) o[f] = zero8();
    float rmax[8], lsum[8];
#pragma unroll
    for (int v = 0; v < 8; ++v) { rmax[v] = -3.0e38f; lsum[v] = 0.0f; }

    for (int kt = 0; kt < S_; kt += 32) {
        __syncthreads();
        // K tile: 32 rows x 96 hd, rows contiguous -> single linear b128 copy
        {
            const uint4* src = reinterpret_cast<const uint4*>(Kp + (size_t)kt * HD_);
            uint4* dst = reinterpret_cast<uint4*>(Kt);
            for (int i = threadIdx.x; i < (32 * HD_) / 8; i += 256) dst[i] = src[i];
        }
        // V tile: 96 hd-rows x 32 keys; row stride S in global, 4 uint4/row
        {
            uint4* dst = reinterpret_cast<uint4*>(Vt);
            for (int i = threadIdx.x; i < HD_ * 4; i += 256) {
                int hd = i >> 2, c = i & 3;
                dst[i] = *reinterpret_cast<const uint4*>(
                    Vg + (size_t)hd * S_ + kt + c * 8);
            }
        }
        // prefetch next tiles into cache while we do softmax/WMMA work
        if (kt + 32 < S_) {
            __builtin_prefetch(Kp + (size_t)(kt + 32) * HD_ + threadIdx.x * 12, 0, 1);
            if (threadIdx.x < HD_)
                __builtin_prefetch(Vg + (size_t)threadIdx.x * S_ + kt + 32, 0, 1);
        }
        __syncthreads();

        // S = Q K^T  (two 16-key halves)
        v8f s0 = zero8(), s1 = zero8();
#pragma unroll
        for (int j = 0; j < 3; ++j) {
            v16bf b0 = frag_b(Kt, HD_, 0,  j * 32, lane);
            v16bf b1 = frag_b(Kt, HD_, 16, j * 32, lane);
            s0 = wmma_bf16(aQ[j], b0, s0);
            s1 = wmma_bf16(aQ[j], b1, s1);
        }

        // online softmax (row = (vgpr, lane-half); reduce across 16 lanes)
        float alpha[8];
#pragma unroll
        for (int v = 0; v < 8; ++v) {
            float mx = fmaxf(s0[v], s1[v]);
#pragma unroll
            for (int off = 1; off < 16; off <<= 1)
                mx = fmaxf(mx, __shfl_xor(mx, off, 16));
            float nm = fmaxf(rmax[v], mx);
            float al = __expf(rmax[v] - nm);
            rmax[v] = nm; alpha[v] = al;
            float p0 = __expf(s0[v] - nm);
            float p1 = __expf(s1[v] - nm);
            s0[v] = p0; s1[v] = p1;
            float rs = p0 + p1;
#pragma unroll
            for (int off = 1; off < 16; off <<= 1)
                rs += __shfl_xor(rs, off, 16);
            lsum[v] = lsum[v] * al + rs;
        }
#pragma unroll
        for (int f = 0; f < 6; ++f)
#pragma unroll
            for (int v = 0; v < 8; ++v) o[f][v] *= alpha[v];

        // stage P (C-frag layout -> row-major 16x32 bf16) through LDS
        {
            ushort_t* P = &Pst[wave][0];
            int n = lane & 15, hi = lane >> 4;
#pragma unroll
            for (int v = 0; v < 8; ++v) {
                int m = v + hi * 8;
                P[m * 32 + n]      = f32_to_bf16(s0[v]);
                P[m * 32 + 16 + n] = f32_to_bf16(s1[v]);
            }
        }
        __syncthreads();

        // O += P V  (6 hd-tiles of 16)
        v16bf aP = frag_a(&Pst[wave][0], 32, 0, 0, lane);
#pragma unroll
        for (int f = 0; f < 6; ++f) {
            v16bf bv = frag_b(Vt, 32, f * 16, 0, lane);
            o[f] = wmma_bf16(aP, bv, o[f]);
        }
    }

    // epilogue: normalize and store to [B,S,DIM] bf16 (DIM = H*HD)
    int b = bh >> 4, h = bh & 15;
    int nl = lane & 15, hi = lane >> 4;
#pragma unroll
    for (int f = 0; f < 6; ++f) {
#pragma unroll
        for (int v = 0; v < 8; ++v) {
            int m = q0 + v + hi * 8;
            int col = h * HD_ + f * 16 + nl;
            out[((size_t)b * S_ + m) * DIM_ + col] = f32_to_bf16(o[f][v] / lsum[v]);
        }
    }
}

// ---------------- Kernel 5: output projection GEMM -------------------------
// out_f32[M,DIM] = attn_bf16[M,DIM] @ proj_wb[DIM,DIM]^T + proj_b
// Same 32x64 wave tile + double-buffered A as qkv_gemm.
__global__ void __launch_bounds__(128) proj_gemm(const ushort_t* __restrict__ ab,
                                                 const ushort_t* __restrict__ wb,
                                                 const float* __restrict__ bias,
                                                 float* __restrict__ outp) {
    int mt = blockIdx.x * 32;
    int wave = threadIdx.x >> 5, lane = threadIdx.x & 31;
    int nbase = blockIdx.y * 256 + wave * 64;

    v8f acc0[4], acc1[4];
#pragma unroll
    for (int t = 0; t < 4; ++t) { acc0[t] = zero8(); acc1[t] = zero8(); }

    v16bf a0 = frag_a(ab, DIM_, mt, 0, lane);
    v16bf a1 = frag_a(ab, DIM_, mt + 16, 0, lane);
    for (int k0 = 0; k0 < DIM_; k0 += 32) {
        v16bf b0 = frag_b(wb, DIM_, nbase,      k0, lane);
        v16bf b1 = frag_b(wb, DIM_, nbase + 16, k0, lane);
        v16bf b2 = frag_b(wb, DIM_, nbase + 32, k0, lane);
        v16bf b3 = frag_b(wb, DIM_, nbase + 48, k0, lane);
        int kn = (k0 + 32 < DIM_) ? (k0 + 32) : 0;
        v16bf a0n = frag_a(ab, DIM_, mt,      kn, lane);
        v16bf a1n = frag_a(ab, DIM_, mt + 16, kn, lane);
        acc0[0] = wmma_bf16(a0, b0, acc0[0]);
        acc1[0] = wmma_bf16(a1, b0, acc1[0]);
        acc0[1] = wmma_bf16(a0, b1, acc0[1]);
        acc1[1] = wmma_bf16(a1, b1, acc1[1]);
        acc0[2] = wmma_bf16(a0, b2, acc0[2]);
        acc1[2] = wmma_bf16(a1, b2, acc1[2]);
        acc0[3] = wmma_bf16(a0, b3, acc0[3]);
        acc1[3] = wmma_bf16(a1, b3, acc1[3]);
        a0 = a0n; a1 = a1n;
    }
    int nl = lane & 15, hi = lane >> 4;
#pragma unroll
    for (int ms = 0; ms < 2; ++ms) {
        v8f* acc = ms ? acc1 : acc0;
#pragma unroll
        for (int t = 0; t < 4; ++t) {
            int n = nbase + t * 16 + nl;
            float bs = bias[n];
#pragma unroll
            for (int v = 0; v < 8; ++v) {
                int m = mt + ms * 16 + v + hi * 8;
                outp[(size_t)m * DIM_ + n] = acc[t][v] + bs;
            }
        }
    }
}

// ---------------- host launch ----------------------------------------------
extern "C" void kernel_launch(void* const* d_in, const int* in_sizes, int n_in,
                              void* d_out, int out_size, void* d_ws, size_t ws_size,
                              hipStream_t stream) {
    (void)in_sizes; (void)n_in; (void)out_size; (void)ws_size;
    const float* x       = (const float*)d_in[0];
    const float* coords  = (const float*)d_in[1];
    const float* qkv_w   = (const float*)d_in[2];
    const float* qkv_b   = (const float*)d_in[3];
    const float* proj_w  = (const float*)d_in[4];
    const float* proj_b  = (const float*)d_in[5];
    const float* mixf    = (const float*)d_in[6];
    float* outp = (float*)d_out;

    // workspace layout (bytes)
    char* ws = (char*)d_ws;
    ushort_t* xb      = (ushort_t*)(ws);                          // 4096*1536
    ushort_t* qkv_wb  = (ushort_t*)(ws + 12582912);               // 4608*1536
    ushort_t* proj_wb = (ushort_t*)(ws + 26738688);               // 1536*1536
    ushort_t* qkv     = (ushort_t*)(ws + 31457280);               // 3*2*16*2048*96
    ushort_t* attn_o  = (ushort_t*)(ws + 69206016);               // 4096*1536

    // 1) conversions
    {
        int n1 = M_ * DIM_;       // x
        int n2 = QKVN_ * DIM_;    // qkv_w
        int n3 = DIM_ * DIM_;     // proj_w
        convert_f32_bf16<<<(n1 + 255) / 256, 256, 0, stream>>>(x, xb, n1);
        convert_f32_bf16<<<(n2 + 255) / 256, 256, 0, stream>>>(qkv_w, qkv_wb, n2);
        convert_f32_bf16<<<(n3 + 255) / 256, 256, 0, stream>>>(proj_w, proj_wb, n3);
    }
    // 2) QKV GEMM: M=4096, N=4608 (writes Q,K row-major; V transposed)
    qkv_gemm<<<dim3(M_ / 32, QKVN_ / 256), 128, 0, stream>>>(xb, qkv_wb, qkv_b, qkv);
    // 3) RoPE on q,k
    {
        int total = B_ * H_ * S_ * NPAIR_;
        rope_kernel<<<(total + 255) / 256, 256, 0, stream>>>(qkv, coords, mixf);
    }
    // 4) flash attention
    {
        const ushort_t* Qp = qkv;
        const ushort_t* Kp = qkv + (size_t)B_ * H_ * S_ * HD_;
        const ushort_t* Vp = qkv + (size_t)2 * B_ * H_ * S_ * HD_;
        flash_attn<<<dim3(B_ * H_ * (S_ / 128)), 256, 0, stream>>>(Qp, Kp, Vp, attn_o);
    }
    // 5) projection
    proj_gemm<<<dim3(M_ / 32, DIM_ / 256), 128, 0, stream>>>(attn_o, proj_wb, proj_b, outp);
}
